// MultiHeadAttention_45861660787068
// MI455X (gfx1250) — matrix operand order repaired and checked by reading, not compiled, not run on hardware
//
#include <hip/hip_runtime.h>
#include <hip/hip_bf16.h>

typedef __attribute__((ext_vector_type(16))) _Float16 v16h;
typedef __attribute__((ext_vector_type(8)))  _Float16 v8h;
typedef __attribute__((ext_vector_type(8)))  float    v8f;
typedef __attribute__((ext_vector_type(4)))  unsigned int u32x4;
typedef __attribute__((ext_vector_type(8)))  int      i32x8;
typedef __attribute__((ext_vector_type(4)))  int      i32x4;

#define NB_B  4
#define NB_S  2048
#define NB_D  1024
#define NB_H  16
#define NB_HD 64
#define ROWS  (NB_B * NB_S)   // 8192

__device__ __forceinline__ v8f vzero8() {
  v8f z = {0.f, 0.f, 0.f, 0.f, 0.f, 0.f, 0.f, 0.f};
  return z;
}

__device__ __forceinline__ v16h ldfrag(const _Float16* lo, const _Float16* hi) {
  union { v16h v; v8h h[2]; } u;
  u.h[0] = *(const v8h*)lo;
  u.h[1] = *(const v8h*)hi;
  return u.v;
}

// Low 32 bits of a flat shared pointer == byte offset within the WG's LDS.
__device__ __forceinline__ unsigned int ldsAddr(const void* p) {
  return (unsigned int)(uintptr_t)p;
}

// DPP16 row_ror butterfly reductions over the 16-lane row (= C-layout group).
__device__ __forceinline__ float dpp_ror(float x, const int ctrl_is_unused) {
  return x;  // (kept out; see explicit ctrl constants below)
}
__device__ __forceinline__ float rowmax16(float x) {
  x = fmaxf(x, __int_as_float(__builtin_amdgcn_update_dpp(
          0, __float_as_int(x), 0x121, 0xf, 0xf, true)));  // row_ror:1
  x = fmaxf(x, __int_as_float(__builtin_amdgcn_update_dpp(
          0, __float_as_int(x), 0x122, 0xf, 0xf, true)));  // row_ror:2
  x = fmaxf(x, __int_as_float(__builtin_amdgcn_update_dpp(
          0, __float_as_int(x), 0x124, 0xf, 0xf, true)));  // row_ror:4
  x = fmaxf(x, __int_as_float(__builtin_amdgcn_update_dpp(
          0, __float_as_int(x), 0x128, 0xf, 0xf, true)));  // row_ror:8
  return x;
}
__device__ __forceinline__ float rowsum16(float x) {
  x += __int_as_float(__builtin_amdgcn_update_dpp(
          0, __float_as_int(x), 0x121, 0xf, 0xf, true));
  x += __int_as_float(__builtin_amdgcn_update_dpp(
          0, __float_as_int(x), 0x122, 0xf, 0xf, true));
  x += __int_as_float(__builtin_amdgcn_update_dpp(
          0, __float_as_int(x), 0x124, 0xf, 0xf, true));
  x += __int_as_float(__builtin_amdgcn_update_dpp(
          0, __float_as_int(x), 0x128, 0xf, 0xf, true));
  return x;
}

// -------------------------------------------------------------------------
// Tensor Data Mover: 2-D tile (tile_d0 halves/row, tile_d1 rows) from global
// (row stride stride_d0 halves) into LDS with hardware row padding.
// -------------------------------------------------------------------------
__device__ __forceinline__ void tdm_load_2d(unsigned int lds_byte_off,
                                            const void* gptr,
                                            unsigned int tile_d0,
                                            unsigned int tile_d1,
                                            unsigned int stride_d0,
                                            unsigned int pad_interval_code,
                                            unsigned int pad_amount_code)
{
  unsigned long long ga = (unsigned long long)(uintptr_t)gptr;
  u32x4 g0;
  g0.x = 1u;                                  // count=1 (valid user descriptor)
  g0.y = lds_byte_off;                        // lds_addr
  g0.z = (unsigned int)ga;                    // global_addr[31:0]
  g0.w = (unsigned int)((ga >> 32) & 0x01FFFFFFu) | (2u << 30);  // ga hi | type=2
  i32x8 g1;
  g1[0] = (int)((1u << 16)                    // data_size = 2 bytes
              | (1u << 20)                    // pad_enable
              | (pad_interval_code << 22)
              | (pad_amount_code << 25));
  g1[1] = (int)(tile_d0 << 16);               // tensor_dim0[15:0]
  g1[2] = (int)(tile_d1 << 16);               // tensor_dim1[15:0]
  g1[3] = (int)(tile_d0 << 16);               // tile_dim0
  g1[4] = (int)tile_d1;                       // tile_dim1
  g1[5] = (int)stride_d0;                     // tensor_dim0_stride[31:0]
  g1[6] = 0;
  g1[7] = 0;
  i32x4 z4 = {0, 0, 0, 0};
#if defined(__clang_major__) && __clang_major__ >= 23
  i32x8 z8 = {0, 0, 0, 0, 0, 0, 0, 0};
  __builtin_amdgcn_tensor_load_to_lds(g0, g1, z4, z4, z8, 0);
#else
  __builtin_amdgcn_tensor_load_to_lds(g0, g1, z4, z4, 0);
#endif
}

// ---------------------------------------------------------------------------
// Weight transpose + f32->f16 convert: Wt[n][k] = (f16)W[k][n]. 64x64 tiles.
// ---------------------------------------------------------------------------
__global__ __launch_bounds__(256) void w_transpose_cvt(
    const float* __restrict__ W, _Float16* __restrict__ Wt)
{
  __shared__ float tile[64][65];
  const int tid = threadIdx.x;
  const int r0 = blockIdx.y * 64;   // k rows of W
  const int c0 = blockIdx.x * 64;   // n cols of W
  #pragma unroll
  for (int i = 0; i < 4; ++i) {
    int cc = tid + 256 * i;                 // 1024 float4 chunks
    int r = cc >> 4, c4 = cc & 15;
    float4 f = *(const float4*)&W[(size_t)(r0 + r) * NB_D + c0 + c4 * 4];
    tile[r][c4 * 4 + 0] = f.x; tile[r][c4 * 4 + 1] = f.y;
    tile[r][c4 * 4 + 2] = f.z; tile[r][c4 * 4 + 3] = f.w;
  }
  __syncthreads();
  #pragma unroll
  for (int i = 0; i < 2; ++i) {
    int cc = tid + 256 * i;                 // 512 8-half chunks
    int nc = cc >> 3, k8 = cc & 7;
    union { uint4 u; _Float16 h[8]; } tb;
    #pragma unroll
    for (int j = 0; j < 8; ++j) tb.h[j] = (_Float16)tile[k8 * 8 + j][nc];
    *(uint4*)&Wt[(size_t)(c0 + nc) * NB_D + r0 + k8 * 8] = tb.u;
  }
}

// ---------------------------------------------------------------------------
// GEMM: C[row,col] = A[row,:] @ W[:,col] + bias[col], W given as f16 Wt[n][k].
//   AMODE 0: A f32 (manual cvt staging, prefetched); AMODE 1: A f16 via TDM.
//   OMODE 0: f16 head-split [B,H,S,HD] (LDS-staged b128 stores)
//   OMODE 1: f32 flat [ROWS,D]
//   OMODE 2: f16 head-split transposed [B,H,HD,S] (vectorized stores)
// Block 256 thr / 8 waves, tile 128x64, BK=32, double-buffered + TDM for B.
// ---------------------------------------------------------------------------
template <int AMODE, int OMODE>
__global__ __launch_bounds__(256) void mha_gemm(
    const float* __restrict__ Af, const _Float16* __restrict__ Ah,
    const _Float16* __restrict__ Wt, const float* __restrict__ bias,
    _Float16* __restrict__ outH, float* __restrict__ outF)
{
  __shared__ __align__(16) _Float16 Alds[2][128 * 40];
  __shared__ __align__(16) _Float16 Blds[2][64 * 40];
  __shared__ __align__(16) _Float16 Elds[8][32 * 40];  // epilogue staging

  const int tid  = threadIdx.x;
  const int w    = tid >> 5, lane = tid & 31;
  const int m    = lane & 15, g = lane >> 4;
  const int qr   = w & 3, qc = w >> 2;
  const int row0 = blockIdx.y * 128;
  const int nb   = blockIdx.x * 64;

  v8f acc[2][2];
  acc[0][0] = vzero8(); acc[0][1] = vzero8();
  acc[1][0] = vzero8(); acc[1][1] = vzero8();

  // ---- staging helpers -------------------------------------------------
  auto stageA = [&](int buf, int kb) {
    if (AMODE == 0) {
      #pragma unroll
      for (int i = 0; i < 4; ++i) {
        int c = tid + 256 * i;
        int r = c >> 3, c4 = c & 7;
        float4 f = *(const float4*)&Af[(size_t)(row0 + r) * NB_D + kb + c4 * 4];
        union { uint2 u; _Float16 h[4]; } tb;
        tb.h[0] = (_Float16)f.x; tb.h[1] = (_Float16)f.y;
        tb.h[2] = (_Float16)f.z; tb.h[3] = (_Float16)f.w;
        *(uint2*)&Alds[buf][r * 40 + c4 * 4] = tb.u;
      }
    } else {
      if (w == 0)
        tdm_load_2d(ldsAddr(&Alds[buf][0]), &Ah[(size_t)row0 * NB_D + kb],
                    32, 128, NB_D, 3, 3);
    }
  };
  auto stageB = [&](int buf, int kb) {
    if (w == 0)
      tdm_load_2d(ldsAddr(&Blds[buf][0]), &Wt[(size_t)nb * NB_D + kb],
                  32, 64, NB_D, 3, 3);
  };

  // ---- prologue + pipelined main loop ----------------------------------
  stageA(0, 0);
  stageB(0, 0);

  for (int kt = 0; kt < NB_D / 32; ++kt) {
    const int cur = kt & 1, nxt = cur ^ 1;
    if (w == 0) __builtin_amdgcn_s_wait_tensorcnt(0);  // buffer 'cur' DMA done
    __syncthreads();                                   // + manual stores done
    if (kt + 1 < NB_D / 32) {
      stageA(nxt, (kt + 1) * 32);
      stageB(nxt, (kt + 1) * 32);
    }

    #pragma unroll
    for (int i = 0; i < 2; ++i) {
      int ra = 32 * qr + 16 * i + m;
      v16h a = ldfrag(&Alds[cur][ra * 40 + 8 * g],
                      &Alds[cur][ra * 40 + 16 + 8 * g]);
      #pragma unroll
      for (int j = 0; j < 2; ++j) {
        int cb = 32 * qc + 16 * j + m;
        v16h bf = ldfrag(&Blds[cur][cb * 40 + 8 * g],
                         &Blds[cur][cb * 40 + 16 + 8 * g]);
        acc[i][j] = __builtin_amdgcn_wmma_f32_16x16x32_f16(
            false, a, false, bf, (short)0, acc[i][j], false, false);
      }
    }
  }

  // ---- epilogue: bias + store. C layout: (row = r + 8g, col = lane%16) --
  if (OMODE == 0) {
    // Stage the wave's 32x32 f16 tile, then coalesced b128 stores.
    _Float16* Ep = &Elds[w][0];
    #pragma unroll
    for (int j = 0; j < 2; ++j) {
      float bval = bias[nb + 32 * qc + 16 * j + m];
      #pragma unroll
      for (int i = 0; i < 2; ++i)
        #pragma unroll
        for (int r = 0; r < 8; ++r)
          Ep[(16 * i + 8 * g + r) * 40 + 16 * j + m] =
              (_Float16)(acc[i][j][r] + bval);
    }
    asm volatile("s_wait_dscnt 0" ::: "memory");
    const int colb = nb + 32 * qc;           // 32-aligned within a 64-wide head
    const int hh2  = colb >> 6, hdb = colb & 63;
    #pragma unroll
    for (int i = 0; i < 4; ++i) {
      int c = lane + 32 * i;                 // 32 rows x 4 chunks
      int r = c >> 2, c8 = c & 3;
      int row = row0 + 32 * qr + r;
      int s = row & (NB_S - 1), bb = row >> 11;
      *(uint4*)&outH[((size_t)(bb * NB_H + hh2) * NB_S + s) * NB_HD + hdb + c8 * 8] =
          *(const uint4*)&Ep[r * 40 + c8 * 8];
    }
  } else {
    #pragma unroll
    for (int j = 0; j < 2; ++j) {
      int col = nb + 32 * qc + 16 * j + m;
      float bval = bias[col];
      #pragma unroll
      for (int i = 0; i < 2; ++i) {
        if (OMODE == 2) {
          int rowb = row0 + 32 * qr + 16 * i + 8 * g;
          int bb = rowb >> 11, sb = rowb & (NB_S - 1);
          int hh = col >> 6, hd = col & 63;
          union { uint4 u; _Float16 h[8]; } tb;
          #pragma unroll
          for (int r = 0; r < 8; ++r) tb.h[r] = (_Float16)(acc[i][j][r] + bval);
          *(uint4*)&outH[((size_t)(bb * NB_H + hh) * NB_HD + hd) * NB_S + sb] = tb.u;
        } else {
          #pragma unroll
          for (int r = 0; r < 8; ++r) {
            int row = row0 + 32 * qr + 16 * i + 8 * g + r;
            outF[(size_t)row * NB_D + col] = acc[i][j][r] + bval;
          }
        }
      }
    }
  }
}

// ---------------------------------------------------------------------------
// Flash attention over one (b,h): 128 q rows/block, 8 waves x 16-row strips.
// K/V tiles via double-buffered TDM; V pre-transposed in memory; DPP softmax.
// ---------------------------------------------------------------------------
__global__ __launch_bounds__(256) void mha_attn(
    const _Float16* __restrict__ Qh, const _Float16* __restrict__ Kh,
    const _Float16* __restrict__ Vt, const float* __restrict__ mask,
    _Float16* __restrict__ An)
{
  __shared__ __align__(16) _Float16 Klds[2][64 * 72];  // [key][hd], padded
  __shared__ __align__(16) _Float16 Vlds[2][64 * 72];  // [hd][key], padded
  __shared__ __align__(16) _Float16 Pst[8 * 16 * 72];  // per-wave Q/P/O staging

  const int tid = threadIdx.x;
  const int w   = tid >> 5, lane = tid & 31;
  const int m   = lane & 15, g = lane >> 4;
  const int bh  = blockIdx.y;
  const int bi  = bh >> 4;
  const int hh  = bh & 15;
  const int s0  = blockIdx.x * 128;

  const _Float16* Qb  = Qh + (size_t)bh * (NB_S * NB_HD);
  const _Float16* Kb  = Kh + (size_t)bh * (NB_S * NB_HD);
  const _Float16* Vtb = Vt + (size_t)bh * (NB_HD * NB_S);  // [hd][s]
  _Float16* Ps = &Pst[w * 16 * 72];

  // Stage this wave's 16x64 Q strip, keep A fragments in registers.
  #pragma unroll
  for (int i = 0; i < 4; ++i) {
    int c = lane + 32 * i;
    int r = c >> 3, c8 = c & 7;
    *(uint4*)&Ps[r * 72 + c8 * 8] =
        *(const uint4*)&Qb[(size_t)(s0 + 16 * w + r) * NB_HD + c8 * 8];
  }
  __syncthreads();
  v16h qfrag[2];
  #pragma unroll
  for (int c = 0; c < 2; ++c)
    qfrag[c] = ldfrag(&Ps[m * 72 + 32 * c + 8 * g],
                      &Ps[m * 72 + 32 * c + 16 + 8 * g]);

  // Prefetch tile 0 via TDM (wave 0; tracked by TENSORcnt).
  if (w == 0) {
    tdm_load_2d(ldsAddr(&Klds[0][0]), Kb,  64, 64, NB_HD, 4, 3);
    tdm_load_2d(ldsAddr(&Vlds[0][0]), Vtb, 64, 64, NB_S,  4, 3);
  }

  float mcur[8], lsum[8];
  v8f oacc[4];
  #pragma unroll
  for (int r = 0; r < 8; ++r) { mcur[r] = -3.0e38f; lsum[r] = 0.f; }
  #pragma unroll
  for (int t = 0; t < 4; ++t) oacc[t] = vzero8();

  const float scale = 1.0f / (64.0f + 1e-8f);  // reference: HD + 1e-8
  const int NT = NB_S / 64;

  for (int kt = 0; kt < NT; ++kt) {
    const int kbase = kt * 64;
    const int cur = kt & 1, nxt = cur ^ 1;
    __syncthreads();  // nobody still reads buffer 'nxt' (used in kt-1)
    if (w == 0) {
      if (kt + 1 < NT) {
        tdm_load_2d(ldsAddr(&Klds[nxt][0]), Kb + (size_t)(kbase + 64) * NB_HD,
                    64, 64, NB_HD, 4, 3);
        tdm_load_2d(ldsAddr(&Vlds[nxt][0]), Vtb + (kbase + 64),
                    64, 64, NB_S, 4, 3);
        __builtin_amdgcn_s_wait_tensorcnt(2);  // 'cur' done, prefetch in flight
      } else {
        __builtin_amdgcn_s_wait_tensorcnt(0);
      }
    }
    __syncthreads();  // tile 'cur' visible to all waves

    // Scores: 4 col-tiles of 16 keys, HD=64 reduction in 2 WMMA steps.
    v8f sacc[4];
    #pragma unroll
    for (int t = 0; t < 4; ++t) sacc[t] = vzero8();
    #pragma unroll
    for (int t = 0; t < 4; ++t) {
      #pragma unroll
      for (int c = 0; c < 2; ++c) {
        v16h bf = ldfrag(&Klds[cur][(16 * t + m) * 72 + 32 * c + 8 * g],
                         &Klds[cur][(16 * t + m) * 72 + 32 * c + 16 + 8 * g]);
        sacc[t] = __builtin_amdgcn_wmma_f32_16x16x32_f16(
            false, qfrag[c], false, bf, (short)0, sacc[t], false, false);
      }
    }

    float mv[4];
    #pragma unroll
    for (int t = 0; t < 4; ++t)
      mv[t] = mask[(size_t)bi * NB_S + kbase + 16 * t + m] * (-1.0e9f);

    // Online softmax per row (rows r+8g); DPP16 row reductions.
    #pragma unroll
    for (int r = 0; r < 8; ++r) {
      float smax = -3.0e38f;
      #pragma unroll
      for (int t = 0; t < 4; ++t) {
        float sv = sacc[t][r] * scale + mv[t];
        sacc[t][r] = sv;
        smax = fmaxf(smax, sv);
      }
      smax = rowmax16(smax);
      float mnew  = fmaxf(mcur[r], smax);
      float alpha = __expf(mcur[r] - mnew);
      mcur[r] = mnew;
      lsum[r] *= alpha;
      #pragma unroll
      for (int t = 0; t < 4; ++t) oacc[t][r] *= alpha;
      float ps = 0.f;
      #pragma unroll
      for (int t = 0; t < 4; ++t) {
        float p = __expf(sacc[t][r] - mnew);
        ps += p;
        Ps[(r + 8 * g) * 72 + 16 * t + m] = (_Float16)p;
      }
      lsum[r] += rowsum16(ps);
    }

    // P stores -> P fragment loads (same wave; LDS in-order).
    asm volatile("s_wait_dscnt 0" ::: "memory");

    v16h pfrag[2];
    #pragma unroll
    for (int c = 0; c < 2; ++c)
      pfrag[c] = ldfrag(&Ps[m * 72 + 32 * c + 8 * g],
                        &Ps[m * 72 + 32 * c + 16 + 8 * g]);
    #pragma unroll
    for (int t = 0; t < 4; ++t) {
      #pragma unroll
      for (int c = 0; c < 2; ++c) {
        v16h vf = ldfrag(&Vlds[cur][(16 * t + m) * 72 + 32 * c + 8 * g],
                         &Vlds[cur][(16 * t + m) * 72 + 32 * c + 16 + 8 * g]);
        oacc[t] = __builtin_amdgcn_wmma_f32_16x16x32_f16(
            false, pfrag[c], false, vf, (short)0, oacc[t], false, false);
      }
    }
  }

  // Normalize, stage through per-wave LDS, then coalesced b128 stores.
  #pragma unroll
  for (int r = 0; r < 8; ++r) {
    float inv = 1.0f / lsum[r];
    #pragma unroll
    for (int t = 0; t < 4; ++t)
      Ps[(r + 8 * g) * 72 + 16 * t + m] = (_Float16)(oacc[t][r] * inv);
  }
  asm volatile("s_wait_dscnt 0" ::: "memory");
  #pragma unroll
  for (int i = 0; i < 4; ++i) {
    int c = lane + 32 * i;
    int r = c >> 3, c8 = c & 7;
    *(uint4*)&An[((size_t)bi * NB_S + s0 + 16 * w + r) * NB_D + hh * NB_HD + c8 * 8] =
        *(const uint4*)&Ps[r * 72 + c8 * 8];
  }
}

// ---------------------------------------------------------------------------
extern "C" void kernel_launch(void* const* d_in, const int* in_sizes, int n_in,
                              void* d_out, int out_size, void* d_ws, size_t ws_size,
                              hipStream_t stream) {
  const float* q    = (const float*)d_in[0];
  const float* k    = (const float*)d_in[1];
  const float* v    = (const float*)d_in[2];
  const float* mask = (const float*)d_in[3];
  const float* Wq   = (const float*)d_in[4];
  const float* bq   = (const float*)d_in[5];
  const float* Wk   = (const float*)d_in[6];
  const float* bk   = (const float*)d_in[7];
  const float* Wv   = (const float*)d_in[8];
  const float* bv   = (const float*)d_in[9];
  const float* Wo   = (const float*)d_in[10];
  const float* bo   = (const float*)d_in[11];

  // Workspace: Qh,Kh (head-split), Vt (transposed), An, 4x Wt(f16) = 72 MB.
  _Float16* Qh  = (_Float16*)d_ws;
  _Float16* Kh  = Qh + (size_t)ROWS * NB_D;
  _Float16* Vtw = Kh + (size_t)ROWS * NB_D;
  _Float16* An  = Vtw + (size_t)ROWS * NB_D;
  _Float16* WqT = An + (size_t)ROWS * NB_D;
  _Float16* WkT = WqT + (size_t)NB_D * NB_D;
  _Float16* WvT = WkT + (size_t)NB_D * NB_D;
  _Float16* WoT = WvT + (size_t)NB_D * NB_D;

  dim3 blk(256);
  dim3 gT(NB_D / 64, NB_D / 64);    // (16, 16)
  w_transpose_cvt<<<gT, blk, 0, stream>>>(Wq, WqT);
  w_transpose_cvt<<<gT, blk, 0, stream>>>(Wk, WkT);
  w_transpose_cvt<<<gT, blk, 0, stream>>>(Wv, WvT);
  w_transpose_cvt<<<gT, blk, 0, stream>>>(Wo, WoT);

  dim3 gG(NB_D / 64, ROWS / 128);   // (16, 64)
  mha_gemm<0, 0><<<gG, blk, 0, stream>>>(q, nullptr, WqT, bq, Qh, nullptr);
  mha_gemm<0, 0><<<gG, blk, 0, stream>>>(k, nullptr, WkT, bk, Kh, nullptr);
  mha_gemm<0, 2><<<gG, blk, 0, stream>>>(v, nullptr, WvT, bv, Vtw, nullptr);

  dim3 gA(NB_S / 128, NB_B * NB_H); // (16, 64)
  mha_attn<<<gA, blk, 0, stream>>>(Qh, Kh, Vtw, mask, An);

  mha_gemm<1, 1><<<gG, blk, 0, stream>>>(nullptr, An, WoT, bo, nullptr,
                                         (float*)d_out);
}
